// SelfAttentionBlock_50646254355141
// MI455X (gfx1250) — matrix operand order repaired and checked by reading, not compiled, not run on hardware
//
#include <hip/hip_runtime.h>
#include <hip/hip_bf16.h>

#define NPIX 9216
#define MTOT 18432   // B * NPIX
#define CIN  512
#define CKD  512
#define CVD  256
#define COD  512

typedef __attribute__((ext_vector_type(16))) __bf16 v16bf;
typedef __attribute__((ext_vector_type(8)))  float  v8f;
typedef __attribute__((ext_vector_type(4)))  unsigned int v4u;
typedef __attribute__((ext_vector_type(8)))  int v8i;
typedef __attribute__((ext_vector_type(4)))  int v4i;

union AF { v16bf v; uint4 q[2]; };

// ---------------------------------------------------------------------------
// Tensor Data Mover: DMA a 2D tile (16-bit elements) from global to LDS.
// Descriptor per CDNA5 ISA:
//  G0: [1:0]=count=1, [63:32]=lds_addr, [120:64]=global_addr, [127:126]=type=2
//  G1: [17:16]=data_size(1=2B), [79:48]=tensor_dim0, [111:80]=tensor_dim1,
//      [127:112]=tile_dim0, [143:128]=tile_dim1, [207:160]=tensor_dim0_stride
//  G2/G3: zero (2D). Completion tracked with TENSORcnt.
// ---------------------------------------------------------------------------
__device__ inline void tdm_load_2d(unsigned lds_off, const void* gptr,
                                   unsigned tile_d0, unsigned tile_d1,
                                   unsigned tens_d0, unsigned tens_d1,
                                   unsigned long long stride0) {
    unsigned long long ga = (unsigned long long)(size_t)gptr;
    v4u g0;
    g0[0] = 1u;                                        // count=1, user descriptor
    g0[1] = lds_off;                                   // LDS byte address
    g0[2] = (unsigned)(ga & 0xFFFFFFFFu);              // global_addr[31:0]
    g0[3] = (unsigned)((ga >> 32) & 0x1FFFFFFu) | (2u << 30);  // [56:32] | type=2
    v8i g1;
    g1[0] = (int)(1u << 16);                           // data_size = 2 bytes
    g1[1] = (int)((tens_d0 & 0xFFFFu) << 16);          // tensor_dim0[15:0]
    g1[2] = (int)((tens_d0 >> 16) | ((tens_d1 & 0xFFFFu) << 16));
    g1[3] = (int)((tens_d1 >> 16) | (tile_d0 << 16));  // tile_dim0
    g1[4] = (int)tile_d1;                              // tile_dim1 (tile_dim2=0)
    g1[5] = (int)(stride0 & 0xFFFFFFFFull);            // tensor_dim0_stride[31:0]
    g1[6] = (int)((stride0 >> 32) & 0xFFFFull);        // stride[47:32]
    g1[7] = 0;
    v4i z4 = {0, 0, 0, 0};
#if defined(__clang_major__) && __clang_major__ >= 23
    v8i z8 = {0, 0, 0, 0, 0, 0, 0, 0};
    __builtin_amdgcn_tensor_load_to_lds(g0, g1, z4, z4, z8, 0);
#else
    __builtin_amdgcn_tensor_load_to_lds(g0, g1, z4, z4, 0);
#endif
}

__device__ inline unsigned lds_off_of(const void* p) {
    return (unsigned)(size_t)p;   // flat LDS aperture: addr[31:0] = LDS offset
}

// ---------------------------------------------------------------------------
// WMMA fragment loaders (gfx1250 wave32, 16x16x32 bf16 -> f32)
// A frag (16x32): lanes 0-15 row=lane, K {0..7,16..23}; lanes 16-31 row=lane-16,
//                 K {8..15,24..31}. Tile is row-major [row][k], leading dim ldk.
// B frag (32x16): lanes 0-15 col=lane K {0..7,16..23}; lanes 16-31 col=lane-16 K {8..15,24..31}.
//                 Source is B^T row-major [ncol][k].
// ---------------------------------------------------------------------------
__device__ inline v16bf load_afrag(const __bf16* base, int row0, int ldk, int koff) {
    int lane = threadIdx.x & 31;
    int m  = row0 + (lane & 15);
    int kb = (lane >> 4) ? 8 : 0;
    AF f;
    f.q[0] = *(const uint4*)(base + (size_t)m * ldk + koff + kb);
    f.q[1] = *(const uint4*)(base + (size_t)m * ldk + koff + kb + 16);
    return f.v;
}

__device__ inline v16bf load_bfrag(const __bf16* base, int ncol0, int ldk, int koff) {
    int lane = threadIdx.x & 31;
    int n  = ncol0 + (lane & 15);
    int kb = (lane >> 4) ? 8 : 0;
    AF f;
    f.q[0] = *(const uint4*)(base + (size_t)n * ldk + koff + kb);
    f.q[1] = *(const uint4*)(base + (size_t)n * ldk + koff + kb + 16);
    return f.v;
}

__device__ inline v8f wmma_bf16(v16bf a, v16bf b, v8f c) {
    return __builtin_amdgcn_wmma_f32_16x16x32_bf16(false, a, false, b, (short)0, c,
                                                   false, false);
}

// ---------------------------------------------------------------------------
// Small prep kernels
// ---------------------------------------------------------------------------
__global__ void cvt_bf16_kernel(const float* __restrict__ src,
                                __bf16* __restrict__ dst, int n) {
    int i = blockIdx.x * 256 + threadIdx.x;
    if (i < n) dst[i] = (__bf16)src[i];
}

__global__ void bnfold_kernel(const float* __restrict__ g, const float* __restrict__ b,
                              const float* __restrict__ m, const float* __restrict__ v,
                              const float* __restrict__ bias,
                              float* __restrict__ alpha, float* __restrict__ beta, int n) {
    int i = blockIdx.x * 256 + threadIdx.x;
    if (i < n) {
        float s = g[i] * rsqrtf(v[i] + 1e-5f);
        alpha[i] = s;
        beta[i]  = s * bias[i] + b[i] - m[i] * s;
    }
}

// x [B][C][N] f32 (c-major) -> xT [B*N][C] bf16 (pixel-major)
__global__ void __launch_bounds__(256)
transpose_x_kernel(const float* __restrict__ x, __bf16* __restrict__ xT) {
    __shared__ float tile[32][33];
    int b  = blockIdx.z;
    int c0 = blockIdx.y * 32;
    int n0 = blockIdx.x * 32;
    int tx = threadIdx.x & 31, ty = threadIdx.x >> 5;
#pragma unroll
    for (int i = 0; i < 32; i += 8)
        tile[ty + i][tx] = x[((size_t)(b * CIN + c0 + ty + i)) * NPIX + n0 + tx];
    __syncthreads();
#pragma unroll
    for (int i = 0; i < 32; i += 8)
        xT[((size_t)(b * NPIX + n0 + ty + i)) * CIN + c0 + tx] = (__bf16)tile[tx][ty + i];
}

// ---------------------------------------------------------------------------
// bf16 WMMA GEMM, TDM double-buffered: out[m][n] = epi( sum_k A[m][k]*Bt[n][k] )
// 128x64 tile / WG (4 waves); each wave: 32 rows x 64 cols = 8 WMMAs per k-step.
// All fragment loads for a k-step are issued before the WMMA chain so the
// scheduler can overlap LDS latency with matrix ops (partial dscnt waits).
// EPI 0: acc + beta[n]            -> bf16, n-major  [M][N]
// EPI 1: relu(alpha[n]*acc+beta)  -> bf16, n-major
// EPI 2: acc + beta[n]            -> bf16, c-major  [b][n][pix]
// EPI 3: relu(alpha[n]*acc+beta)  -> f32,  c-major  (final output)
// ---------------------------------------------------------------------------
template <int EPI>
__global__ void __launch_bounds__(128)
gemm_bf16_kernel(const __bf16* __restrict__ A, const __bf16* __restrict__ Bt,
                 const float* __restrict__ alpha, const float* __restrict__ beta,
                 int M, int N, int K,
                 __bf16* __restrict__ outb, float* __restrict__ outf) {
    __shared__ __bf16 Al[2][128 * 32];   // 16 KB
    __shared__ __bf16 Bl[2][64 * 32];    //  8 KB

    int tm = blockIdx.x * 128;
    int tn = blockIdx.y * 64;
    int t = threadIdx.x, wave = t >> 5, lane = t & 31;
    int nk = K >> 5;

    if (wave == 0) {   // prologue: DMA first k-chunk
        tdm_load_2d(lds_off_of(&Al[0][0]), A + (size_t)tm * K, 32, 128, 32, 128, K);
        tdm_load_2d(lds_off_of(&Bl[0][0]), Bt + (size_t)tn * K, 32, 64, 32, 64, K);
    }

    v8f acc[2][4];
#pragma unroll
    for (int s = 0; s < 2; ++s)
#pragma unroll
        for (int f = 0; f < 4; ++f) acc[s][f] = (v8f){0, 0, 0, 0, 0, 0, 0, 0};

    for (int kt = 0; kt < nk; ++kt) {
        if (wave == 0) __builtin_amdgcn_s_wait_tensorcnt(0);
        __syncthreads();                       // tile kt visible to all waves
        if (wave == 0 && kt + 1 < nk) {        // DMA tile kt+1 into other buffer
            int nb = (kt + 1) & 1;
            tdm_load_2d(lds_off_of(&Al[nb][0]),
                        A + (size_t)tm * K + (kt + 1) * 32, 32, 128, 32, 128, K);
            tdm_load_2d(lds_off_of(&Bl[nb][0]),
                        Bt + (size_t)tn * K + (kt + 1) * 32, 32, 64, 32, 64, K);
        }
        const __bf16* Ac = Al[kt & 1];
        const __bf16* Bc = Bl[kt & 1];
        // Issue ALL fragment loads, then the WMMA chain (independent loads).
        v16bf a0 = load_afrag(Ac, wave * 32, 32, 0);
        v16bf a1 = load_afrag(Ac, wave * 32 + 16, 32, 0);
        v16bf bfr[4];
#pragma unroll
        for (int f = 0; f < 4; ++f) bfr[f] = load_bfrag(Bc, f * 16, 32, 0);
#pragma unroll
        for (int f = 0; f < 4; ++f) {
            acc[0][f] = wmma_bf16(a0, bfr[f], acc[0][f]);
            acc[1][f] = wmma_bf16(a1, bfr[f], acc[1][f]);
        }
    }

    // Epilogue. C-frag: VGPR r -> row (strip + r + 8*(lane>=16)), col lane&15.
    int rb = (lane >> 4) ? 8 : 0;
    int ncl = lane & 15;
#pragma unroll
    for (int s = 0; s < 2; ++s) {
#pragma unroll
        for (int f = 0; f < 4; ++f) {
            int n = tn + f * 16 + ncl;
            float al = (EPI == 1 || EPI == 3) ? alpha[n] : 1.0f;
            float be = beta[n];
#pragma unroll
            for (int r = 0; r < 8; ++r) {
                int m = tm + wave * 32 + s * 16 + rb + r;
                float v = acc[s][f][r] * al + be;
                if (EPI == 1 || EPI == 3) v = fmaxf(v, 0.0f);
                if (EPI == 0 || EPI == 1) {
                    outb[(size_t)m * N + n] = (__bf16)v;
                } else {
                    int bb = m / NPIX, pix = m % NPIX;
                    size_t idx = ((size_t)bb * N + n) * NPIX + pix;
                    if (EPI == 2) outb[idx] = (__bf16)v;
                    else          outf[idx] = v;
                }
            }
        }
    }
}

// ---------------------------------------------------------------------------
// Flash attention, TDM double-buffered at the m-tile level.
// WG = 64 query rows (4 waves x 16 rows); ctx accumulator [16 x 256] f32/wave.
// Per m-tile: whole Kk [64][512] and Vt [256][64] are TDM-resident in LDS, so
// the 16 S k-steps and 32 ctx WMMAs run with ONE barrier per tile. The S loop
// is software-pipelined: fragments for step kk+1 load during step kk's WMMAs.
// LDS: Kq 64K + 2xKk 128K + 2xVt 64K + P 8K = 264 KB (fits 320 KB WGP LDS).
// ---------------------------------------------------------------------------
__global__ void __launch_bounds__(128)
attention_kernel(const __bf16* __restrict__ kf, const __bf16* __restrict__ vc,
                 __bf16* __restrict__ ctx) {
    extern __shared__ char smem[];
    __bf16* Kq  = (__bf16*)smem;                         // [64][512]   64 KB
    __bf16* KkB = (__bf16*)(smem + 65536);               // [2][64][512] 128 KB
    __bf16* VtB = (__bf16*)(smem + 65536 + 131072);      // [2][256][64]  64 KB
    __bf16* Pl  = (__bf16*)(smem + 65536 + 131072 + 65536); // [64][64]    8 KB

    int b  = blockIdx.y;
    int q0 = blockIdx.x * 64;
    size_t qrow = (size_t)b * NPIX + q0;
    int t = threadIdx.x, wave = t >> 5, lane = t & 31;

    if (wave == 0) {   // prologue DMA: tile 0
        tdm_load_2d(lds_off_of(KkB), kf + (size_t)b * NPIX * CKD, 512, 64, 512, 64, 512);
        tdm_load_2d(lds_off_of(VtB), vc + (size_t)b * CVD * NPIX, 64, 256, 64, 256, NPIX);
    }
    // Resident Kq tile [64][512] (regular loads, overlaps TDM)
    {
        int row = t >> 1, half = t & 1;
        const uint4* s = (const uint4*)(kf + (qrow + row) * CKD + half * 256);
        uint4* d = (uint4*)(Kq + row * 512 + half * 256);
#pragma unroll
        for (int i = 0; i < 16; ++i) d[i] = s[i];
    }

    const float scale = 0.044194173824159216f;   // 512^-0.5
    float rowmax[8], rowsum[8];
#pragma unroll
    for (int r = 0; r < 8; ++r) { rowmax[r] = -3.0e38f; rowsum[r] = 0.0f; }
    v8f ctxacc[16];
#pragma unroll
    for (int f = 0; f < 16; ++f) ctxacc[f] = (v8f){0, 0, 0, 0, 0, 0, 0, 0};

    for (int it = 0; it < NPIX / 64; ++it) {
        if (wave == 0) __builtin_amdgcn_s_wait_tensorcnt(0);
        __syncthreads();   // tile `it` in LDS; everyone done with tile it-1
        if (wave == 0 && it + 1 < NPIX / 64) {
            int nb = (it + 1) & 1;
            int m1 = (it + 1) * 64;
            tdm_load_2d(lds_off_of(KkB + nb * 64 * 512),
                        kf + ((size_t)b * NPIX + m1) * CKD, 512, 64, 512, 64, 512);
            tdm_load_2d(lds_off_of(VtB + nb * 256 * 64),
                        vc + (size_t)b * CVD * NPIX + m1, 64, 256, 64, 256, NPIX);
        }
        const __bf16* Kk = KkB + (it & 1) * 64 * 512;
        const __bf16* Vt = VtB + (it & 1) * 256 * 64;

        // S = Kq * Kk^T over Ck=512: 16 k-steps, software-pipelined fragments.
        v8f sacc[4];
#pragma unroll
        for (int f = 0; f < 4; ++f) sacc[f] = (v8f){0, 0, 0, 0, 0, 0, 0, 0};

        v16bf a = load_afrag(Kq, wave * 16, 512, 0);
        v16bf bfr[4];
#pragma unroll
        for (int f = 0; f < 4; ++f) bfr[f] = load_bfrag(Kk, f * 16, 512, 0);
#pragma unroll
        for (int kk = 0; kk < 16; ++kk) {
            v16bf a_n;
            v16bf b_n[4];
            if (kk < 15) {   // issue next step's loads before this step's WMMAs
                a_n = load_afrag(Kq, wave * 16, 512, (kk + 1) * 32);
#pragma unroll
                for (int f = 0; f < 4; ++f)
                    b_n[f] = load_bfrag(Kk, f * 16, 512, (kk + 1) * 32);
            }
#pragma unroll
            for (int f = 0; f < 4; ++f) sacc[f] = wmma_bf16(a, bfr[f], sacc[f]);
            if (kk < 15) {
                a = a_n;
#pragma unroll
                for (int f = 0; f < 4; ++f) bfr[f] = b_n[f];
            }
        }

        // Online softmax over this 64-wide m-tile.
        float corr[8];
#pragma unroll
        for (int r = 0; r < 8; ++r) {
            float mx = fmaxf(fmaxf(sacc[0][r], sacc[1][r]),
                             fmaxf(sacc[2][r], sacc[3][r])) * scale;
#pragma unroll
            for (int off = 1; off < 16; off <<= 1)
                mx = fmaxf(mx, __shfl_xor(mx, off, 32));   // 16-lane half reduce
            float nm = fmaxf(rowmax[r], mx);
            corr[r] = __expf(rowmax[r] - nm);
            rowmax[r] = nm;
        }
#pragma unroll
        for (int f = 0; f < 16; ++f)
#pragma unroll
            for (int r = 0; r < 8; ++r) ctxacc[f][r] *= corr[r];

        int prb = (lane >> 4) ? 8 : 0;
        int ncl = lane & 15;
#pragma unroll
        for (int r = 0; r < 8; ++r) {
            float rs = 0.0f;
#pragma unroll
            for (int f = 0; f < 4; ++f) {
                float p = __expf(sacc[f][r] * scale - rowmax[r]);
                rs += p;
                Pl[(wave * 16 + prb + r) * 64 + f * 16 + ncl] = (__bf16)p;
            }
#pragma unroll
            for (int off = 1; off < 16; off <<= 1) rs += __shfl_xor(rs, off, 32);
            rowsum[r] = rowsum[r] * corr[r] + rs;
        }

        // ctx += P[16x64] @ V^T[64x256]; B-fragments batched in groups of 4.
        // (same-wave LDS RAW on Pl: DS ops are in-order within a wave)
#pragma unroll
        for (int kc = 0; kc < 2; ++kc) {
            v16bf ap = load_afrag(Pl, wave * 16, 64, kc * 32);
#pragma unroll
            for (int g = 0; g < 4; ++g) {
                v16bf bv4[4];
#pragma unroll
                for (int f = 0; f < 4; ++f)
                    bv4[f] = load_bfrag(Vt, (g * 4 + f) * 16, 64, kc * 32);
#pragma unroll
                for (int f = 0; f < 4; ++f)
                    ctxacc[g * 4 + f] = wmma_bf16(ap, bv4[f], ctxacc[g * 4 + f]);
            }
        }
    }

    // Finalize: divide by softmax denominator, store bf16 pixel-major [B*N][256]
    int rb = (lane >> 4) ? 8 : 0;
    int ncl = lane & 15;
#pragma unroll
    for (int f = 0; f < 16; ++f) {
#pragma unroll
        for (int r = 0; r < 8; ++r) {
            int m = q0 + wave * 16 + rb + r;
            float v = ctxacc[f][r] / rowsum[r];
            ctx[((size_t)b * NPIX + m) * CVD + f * 16 + ncl] = (__bf16)v;
        }
    }
}

// ---------------------------------------------------------------------------
// Host-side orchestration
// ---------------------------------------------------------------------------
extern "C" void kernel_launch(void* const* d_in, const int* in_sizes, int n_in,
                              void* d_out, int out_size, void* d_ws, size_t ws_size,
                              hipStream_t stream) {
    const float* x  = (const float*)d_in[0];
    const float* wk = (const float*)d_in[1];
    const float* bk = (const float*)d_in[2];
    const float* g1 = (const float*)d_in[3];
    const float* b1 = (const float*)d_in[4];
    const float* m1 = (const float*)d_in[5];
    const float* v1 = (const float*)d_in[6];
    const float* wv = (const float*)d_in[7];
    const float* bv = (const float*)d_in[8];
    const float* wW = (const float*)d_in[9];
    const float* bW = (const float*)d_in[10];
    const float* wo = (const float*)d_in[11];
    const float* bo = (const float*)d_in[12];
    const float* g2 = (const float*)d_in[13];
    const float* b2 = (const float*)d_in[14];
    const float* m2 = (const float*)d_in[15];
    const float* v2 = (const float*)d_in[16];
    float* out = (float*)d_out;

    char* ws = (char*)d_ws;
    __bf16* xT   = (__bf16*)(ws);                  // [18432][512]
    __bf16* kfb  = (__bf16*)(ws + 18874368);       // [18432][512]
    __bf16* vcb  = (__bf16*)(ws + 37748736);       // [2][256][9216]
    __bf16* ctxb = (__bf16*)(ws + 47185920);       // [18432][256]
    __bf16* y1   = xT;                             // reuse: xT dead after GEMM2
    __bf16* wkb  = (__bf16*)(ws + 56623104);       // 512x512
    __bf16* wvb  = (__bf16*)(ws + 57147392);       // 256x512
    __bf16* wWb  = (__bf16*)(ws + 57409536);       // 512x256
    __bf16* wob  = (__bf16*)(ws + 57671680);       // 512x512
    float*  a1   = (float*)(ws + 58195968);
    float*  be1  = a1 + 512;
    float*  a2   = a1 + 1024;
    float*  be2  = a1 + 1536;                      // total ~55.5 MB

    cvt_bf16_kernel<<<(512 * 512 + 255) / 256, 256, 0, stream>>>(wk, wkb, 512 * 512);
    cvt_bf16_kernel<<<(256 * 512 + 255) / 256, 256, 0, stream>>>(wv, wvb, 256 * 512);
    cvt_bf16_kernel<<<(512 * 256 + 255) / 256, 256, 0, stream>>>(wW, wWb, 512 * 256);
    cvt_bf16_kernel<<<(512 * 512 + 255) / 256, 256, 0, stream>>>(wo, wob, 512 * 512);
    bnfold_kernel<<<2, 256, 0, stream>>>(g1, b1, m1, v1, bk, a1, be1, 512);
    bnfold_kernel<<<2, 256, 0, stream>>>(g2, b2, m2, v2, bo, a2, be2, 512);

    transpose_x_kernel<<<dim3(NPIX / 32, CIN / 32, 2), 256, 0, stream>>>(x, xT);

    // k = relu(bn(x @ wk^T))   [18432][512] pixel-major
    gemm_bf16_kernel<1><<<dim3(MTOT / 128, CKD / 64), 128, 0, stream>>>(
        xT, wkb, a1, be1, MTOT, CKD, CIN, kfb, nullptr);
    // val = x @ wv^T + bv      c-major [2][256][9216] (V^T tiles for attention)
    gemm_bf16_kernel<2><<<dim3(MTOT / 128, CVD / 64), 128, 0, stream>>>(
        xT, wvb, nullptr, bv, MTOT, CVD, CIN, vcb, nullptr);

    // ctx = softmax(k^T k * Ck^-0.5) @ v^T   (fused, flash-style, TDM pipelined)
    attention_kernel<<<dim3(NPIX / 64, 2), 128,
                       65536 + 131072 + 65536 + 8192, stream>>>(kfb, vcb, ctxb);

    // y1 = ctx @ wW^T + bW     [18432][512] pixel-major (reuses xT space)
    gemm_bf16_kernel<0><<<dim3(MTOT / 128, COD / 64), 128, 0, stream>>>(
        ctxb, wWb, nullptr, bW, MTOT, COD, CVD, y1, nullptr);
    // out = relu(bn(y1 @ wo^T))  f32 channel-major -> d_out [2][512][96][96]
    gemm_bf16_kernel<3><<<dim3(MTOT / 128, COD / 64), 128, 0, stream>>>(
        y1, wob, a2, be2, MTOT, COD, COD, nullptr, out);
}
